// MultiHeadAttention_53257594470580
// MI455X (gfx1250) — matrix operand order repaired and checked by reading, not compiled, not run on hardware
//
#include <hip/hip_runtime.h>

#define BB 4
#define NN 2048
#define DD 64
#define HH 12
#define HD (HH * DD) /* 768 */

typedef __attribute__((ext_vector_type(16))) __bf16 bf16x16;
typedef __attribute__((ext_vector_type(8))) float f32x8;

union ABv {
    bf16x16 v;
    uint4 q[2];
    unsigned short u[16];
};

// native hardware f32 -> bf16 convert (RNE on gfx1250)
__device__ __forceinline__ unsigned short f2bf(float f) {
    union { __bf16 h; unsigned short u; } c;
    c.h = (__bf16)f;
    return c.u;
}

__device__ __forceinline__ f32x8 wmma_bf16(const ABv& a, const ABv& b, f32x8 c) {
    return __builtin_amdgcn_wmma_f32_16x16x32_bf16(false, a.v, false, b.v,
                                                   (short)0, c, false, false);
}

// ---------------------------------------------------------------------------
// Kernel 1: convert x to bf16; transpose weights into B-operand layouts.
//   WqT/WkT/WvT : [H][dout=64][din=64]   (column-of-output contiguous in K)
//   WoT         : [dout=64][din=768]
// ---------------------------------------------------------------------------
__global__ void prep_kernel(const float* __restrict__ x,
                            const float* __restrict__ Wq,
                            const float* __restrict__ Wk,
                            const float* __restrict__ Wv,
                            const float* __restrict__ Wo,
                            unsigned short* __restrict__ xbf,
                            unsigned short* __restrict__ WqT,
                            unsigned short* __restrict__ WkT,
                            unsigned short* __restrict__ WvT,
                            unsigned short* __restrict__ WoT) {
    int i = blockIdx.x * blockDim.x + threadIdx.x;
    int stride = gridDim.x * blockDim.x;
    for (int t = i; t < BB * NN * DD; t += stride) xbf[t] = f2bf(x[t]);
    for (int t = i; t < HH * DD * DD; t += stride) {
        int h = t / (DD * DD);
        int rem = t % (DD * DD);
        int e = rem / DD;   // dout
        int d = rem % DD;   // din
        int src = h * DD * DD + d * DD + e;
        WqT[t] = f2bf(Wq[src]);
        WkT[t] = f2bf(Wk[src]);
        WvT[t] = f2bf(Wv[src]);
    }
    for (int t = i; t < DD * HD; t += stride) {
        int e = t / HD;     // dout
        int f = t % HD;     // din
        WoT[t] = f2bf(Wo[f * DD + e]);
    }
}

// ---------------------------------------------------------------------------
// Kernel 2: QKV projection.  One wave per (b, h, 16-row query block).
//   Qbf : [B,H,N,D]  (pre-scaled by 1/sqrt(D), bias added)
//   Kbf : [B,H,N,D]
//   Vt  : [B,H,D,N]  (transposed so attn's B-operand loads are contiguous)
// ---------------------------------------------------------------------------
__global__ void __launch_bounds__(256) qkv_kernel(
    const unsigned short* __restrict__ xbf,
    const unsigned short* __restrict__ WqT,
    const unsigned short* __restrict__ WkT,
    const unsigned short* __restrict__ WvT,
    const float* __restrict__ bq, const float* __restrict__ bk,
    const float* __restrict__ bv,
    unsigned short* __restrict__ Qbf, unsigned short* __restrict__ Kbf,
    unsigned short* __restrict__ Vt) {
    const int lane = threadIdx.x & 31;
    const int wid = threadIdx.x >> 5;
    const int nqb = NN / 16;
    int gw = blockIdx.x * 8 + wid;          // global wave id
    int b = gw / (HH * nqb);
    int h = (gw / nqb) % HH;
    int q0 = (gw % nqb) * 16;

    const int nlane = lane & 15;
    const int koff = (lane < 16) ? 0 : 8;   // A-operand K sub-block per half
    const int boff = (lane < 16) ? 0 : 8;   // B-operand K sub-block per half
    const int mrow = (lane < 16) ? 0 : 8;   // C/D row offset per half
    const size_t bh = (size_t)b * HH + h;

    // x A-tiles (16 rows x 32 K), two K-halves covering din 0..63
    const unsigned short* xrow = xbf + ((size_t)b * NN + q0 + nlane) * DD;
    ABv a0, a1;
    a0.q[0] = *(const uint4*)(xrow + koff);
    a0.q[1] = *(const uint4*)(xrow + koff + 16);
    a1.q[0] = *(const uint4*)(xrow + 32 + koff);
    a1.q[1] = *(const uint4*)(xrow + 32 + koff + 16);

    for (int t = 0; t < 4; ++t) {
        int n = t * 16 + nlane;             // output dim handled by this lane
        const unsigned short* wq = WqT + h * DD * DD + n * DD;
        const unsigned short* wk = WkT + h * DD * DD + n * DD;
        const unsigned short* wv = WvT + h * DD * DD + n * DD;
        ABv b0, b1;
        f32x8 c;

        // ---- Q ----
        b0.q[0] = ((const uint4*)(wq + boff))[0];
        b0.q[1] = ((const uint4*)(wq + boff))[2];
        b1.q[0] = ((const uint4*)(wq + 32 + boff))[0];
        b1.q[1] = ((const uint4*)(wq + 32 + boff))[2];
        c = (f32x8){};
        c = wmma_bf16(a0, b0, c);
        c = wmma_bf16(a1, b1, c);
        {
            float bias = bq[h * DD + n];
            for (int r = 0; r < 8; ++r)
                Qbf[(bh * NN + q0 + mrow + r) * DD + n] =
                    f2bf((c[r] + bias) * 0.125f);   // fold 1/sqrt(64)
        }

        // ---- K ----
        b0.q[0] = ((const uint4*)(wk + boff))[0];
        b0.q[1] = ((const uint4*)(wk + boff))[2];
        b1.q[0] = ((const uint4*)(wk + 32 + boff))[0];
        b1.q[1] = ((const uint4*)(wk + 32 + boff))[2];
        c = (f32x8){};
        c = wmma_bf16(a0, b0, c);
        c = wmma_bf16(a1, b1, c);
        {
            float bias = bk[h * DD + n];
            for (int r = 0; r < 8; ++r)
                Kbf[(bh * NN + q0 + mrow + r) * DD + n] = f2bf(c[r] + bias);
        }

        // ---- V (stored transposed: [B,H,D,N]) ----
        b0.q[0] = ((const uint4*)(wv + boff))[0];
        b0.q[1] = ((const uint4*)(wv + boff))[2];
        b1.q[0] = ((const uint4*)(wv + 32 + boff))[0];
        b1.q[1] = ((const uint4*)(wv + 32 + boff))[2];
        c = (f32x8){};
        c = wmma_bf16(a0, b0, c);
        c = wmma_bf16(a1, b1, c);
        {
            float bias = bv[h * DD + n];
            for (int r = 0; r < 8; ++r)
                Vt[(bh * DD + n) * NN + q0 + mrow + r] = f2bf(c[r] + bias);
        }
    }
}

// ---------------------------------------------------------------------------
// Kernel 3: flash attention. One wave per (b, h, 16-query block); loop over
// 64-key blocks with online softmax (wide block amortizes softmax VALU).
// Obf: [B,N,H*D] bf16.
// ---------------------------------------------------------------------------
__global__ void __launch_bounds__(256) attn_kernel(
    const unsigned short* __restrict__ Qbf,
    const unsigned short* __restrict__ Kbf,
    const unsigned short* __restrict__ Vt,
    unsigned short* __restrict__ Obf) {
    __shared__ unsigned short sP[8 * 16 * 64];  // per-wave 16x64 bf16 P tile
    const int lane = threadIdx.x & 31;
    const int wid = threadIdx.x >> 5;
    unsigned short* sPw = sP + wid * (16 * 64);

    const int nqb = NN / 16;
    int gw = blockIdx.x * 8 + wid;
    int b = gw / (HH * nqb);
    int h = (gw / nqb) % HH;
    int q0 = (gw % nqb) * 16;

    const int nlane = lane & 15;
    const int koff = (lane < 16) ? 0 : 8;
    const int boff = (lane < 16) ? 0 : 8;
    const int mrow = (lane < 16) ? 0 : 8;
    const size_t bh = (size_t)b * HH + h;

    // Q A-tiles (already scaled by 1/sqrt(D))
    const unsigned short* qrow = Qbf + (bh * NN + q0 + nlane) * DD;
    ABv qa0, qa1;
    qa0.q[0] = *(const uint4*)(qrow + koff);
    qa0.q[1] = *(const uint4*)(qrow + koff + 16);
    qa1.q[0] = *(const uint4*)(qrow + 32 + koff);
    qa1.q[1] = *(const uint4*)(qrow + 32 + koff + 16);

    float m[8], l[8];
    f32x8 o[4];
    {
        f32x8 z = {};
        for (int t = 0; t < 4; ++t) o[t] = z;
    }
    for (int r = 0; r < 8; ++r) { m[r] = -3.0e38f; l[r] = 0.0f; }

    for (int j = 0; j < NN; j += 64) {
        // ---- scores: S = Q @ K^T for 64 keys (four 16x16 tiles) ----
        f32x8 s[4];
        for (int tt = 0; tt < 4; ++tt) {
            const unsigned short* kr = Kbf + (bh * NN + j + tt * 16 + nlane) * DD;
            ABv kb0, kb1;
            kb0.q[0] = ((const uint4*)(kr + boff))[0];
            kb0.q[1] = ((const uint4*)(kr + boff))[2];
            kb1.q[0] = ((const uint4*)(kr + 32 + boff))[0];
            kb1.q[1] = ((const uint4*)(kr + 32 + boff))[2];
            f32x8 z = {};
            z = wmma_bf16(qa0, kb0, z);
            s[tt] = wmma_bf16(qa1, kb1, z);
        }
        if (j + 64 < NN) {  // prefetch next key block (K and V streams)
            __builtin_prefetch(Kbf + (bh * NN + j + 64 + nlane) * DD, 0, 1);
            __builtin_prefetch(Kbf + (bh * NN + j + 96 + nlane) * DD, 0, 1);
            __builtin_prefetch(Vt + (bh * DD + nlane) * NN + j + 64, 0, 1);
        }

        // ---- online softmax (rows live per-vgpr; 16-lane-half reductions) --
        float mt[8], rs[8], alpha[8], mn[8];
        for (int r = 0; r < 8; ++r)
            mt[r] = fmaxf(fmaxf(s[0][r], s[1][r]), fmaxf(s[2][r], s[3][r]));
        for (int off = 1; off < 16; off <<= 1)
            for (int r = 0; r < 8; ++r)
                mt[r] = fmaxf(mt[r], __shfl_xor(mt[r], off, 32));
        for (int r = 0; r < 8; ++r) {
            mn[r] = fmaxf(m[r], mt[r]);
            alpha[r] = __expf(m[r] - mn[r]);
            m[r] = mn[r];
            rs[r] = 0.0f;
        }
        // exponentiate, stash P tiles into LDS, accumulate row sums
        for (int tt = 0; tt < 4; ++tt) {
            for (int r = 0; r < 8; ++r) {
                float p = __expf(s[tt][r] - mn[r]);
                rs[r] += p;
                sPw[(mrow + r) * 64 + tt * 16 + nlane] = f2bf(p);
            }
        }
        for (int off = 1; off < 16; off <<= 1)
            for (int r = 0; r < 8; ++r) rs[r] += __shfl_xor(rs[r], off, 32);
        for (int r = 0; r < 8; ++r) l[r] = l[r] * alpha[r] + rs[r];
        for (int t = 0; t < 4; ++t)
            for (int r = 0; r < 8; ++r) o[t][r] *= alpha[r];

        asm volatile("s_wait_dscnt 0x0" ::: "memory");

        // P A-operand tiles (keys 0..31 and 32..63 of this block)
        ABv pa0, pa1;
        pa0.q[0] = *(const uint4*)(&sPw[nlane * 64 + koff]);
        pa0.q[1] = *(const uint4*)(&sPw[nlane * 64 + koff + 16]);
        pa1.q[0] = *(const uint4*)(&sPw[nlane * 64 + 32 + koff]);
        pa1.q[1] = *(const uint4*)(&sPw[nlane * 64 + 32 + koff + 16]);

        // ---- O += P @ V  (V transposed => contiguous B-operand loads) ----
        for (int t = 0; t < 4; ++t) {
            const unsigned short* vr = Vt + (bh * DD + t * 16 + nlane) * NN + j;
            ABv vb0, vb1;
            vb0.q[0] = ((const uint4*)(vr + boff))[0];
            vb0.q[1] = ((const uint4*)(vr + boff))[2];
            vb1.q[0] = ((const uint4*)(vr + 32 + boff))[0];
            vb1.q[1] = ((const uint4*)(vr + 32 + boff))[2];
            o[t] = wmma_bf16(pa0, vb0, o[t]);
            o[t] = wmma_bf16(pa1, vb1, o[t]);
        }
    }

    // normalize and store concat-head layout [B,N,H*D]
    for (int t = 0; t < 4; ++t) {
        for (int r = 0; r < 8; ++r) {
            int mr = mrow + r;
            float val = o[t][r] / l[r];
            Obf[((size_t)b * NN + q0 + mr) * HD + h * DD + t * 16 + nlane] =
                f2bf(val);
        }
    }
}

// ---------------------------------------------------------------------------
// Kernel 4: output projection  out[B*N,64] = Obf[B*N,768] @ Wo + bo  (f32 out)
// ---------------------------------------------------------------------------
__global__ void __launch_bounds__(256) proj_kernel(
    const unsigned short* __restrict__ Obf,
    const unsigned short* __restrict__ WoT,
    const float* __restrict__ bo, float* __restrict__ out) {
    const int lane = threadIdx.x & 31;
    const int wid = threadIdx.x >> 5;
    int gw = blockIdx.x * 8 + wid;
    int q0 = gw * 16;   // 16 token-rows per wave across B*N

    const int nlane = lane & 15;
    const int koff = (lane < 16) ? 0 : 8;
    const int boff = (lane < 16) ? 0 : 8;
    const int mrow = (lane < 16) ? 0 : 8;

    f32x8 c[4];
    {
        f32x8 z = {};
        for (int t = 0; t < 4; ++t) c[t] = z;
    }
    const unsigned short* arow = Obf + (size_t)(q0 + nlane) * HD;
    for (int k = 0; k < HD; k += 32) {
        ABv a;
        a.q[0] = *(const uint4*)(arow + k + koff);
        a.q[1] = *(const uint4*)(arow + k + koff + 16);
        for (int t = 0; t < 4; ++t) {
            const unsigned short* w = WoT + (t * 16 + nlane) * HD + k + boff;
            ABv bv;
            bv.q[0] = ((const uint4*)w)[0];
            bv.q[1] = ((const uint4*)w)[2];
            c[t] = wmma_bf16(a, bv, c[t]);
        }
    }
    for (int t = 0; t < 4; ++t) {
        int n = t * 16 + nlane;
        float bias = bo[n];
        for (int r = 0; r < 8; ++r)
            out[(size_t)(q0 + mrow + r) * DD + n] = c[t][r] + bias;
    }
}

// ---------------------------------------------------------------------------
extern "C" void kernel_launch(void* const* d_in, const int* in_sizes, int n_in,
                              void* d_out, int out_size, void* d_ws,
                              size_t ws_size, hipStream_t stream) {
    const float* x = (const float*)d_in[0];
    const float* Wq = (const float*)d_in[1];
    const float* Wk = (const float*)d_in[2];
    const float* Wv = (const float*)d_in[3];
    const float* bq = (const float*)d_in[4];
    const float* bk = (const float*)d_in[5];
    const float* bv = (const float*)d_in[6];
    const float* Wo = (const float*)d_in[7];
    const float* bo = (const float*)d_in[8];
    float* out = (float*)d_out;

    char* ws = (char*)d_ws;
    size_t off = 0;
    auto walloc = [&](size_t bytes) -> void* {
        void* p = ws + off;
        off += (bytes + 255) & ~(size_t)255;
        return p;
    };
    unsigned short* xbf = (unsigned short*)walloc((size_t)BB * NN * DD * 2);
    unsigned short* WqT = (unsigned short*)walloc((size_t)HH * DD * DD * 2);
    unsigned short* WkT = (unsigned short*)walloc((size_t)HH * DD * DD * 2);
    unsigned short* WvT = (unsigned short*)walloc((size_t)HH * DD * DD * 2);
    unsigned short* WoT = (unsigned short*)walloc((size_t)DD * HD * 2);
    unsigned short* Qbf = (unsigned short*)walloc((size_t)BB * HH * NN * DD * 2);
    unsigned short* Kbf = (unsigned short*)walloc((size_t)BB * HH * NN * DD * 2);
    unsigned short* Vt  = (unsigned short*)walloc((size_t)BB * HH * DD * NN * 2);
    unsigned short* Obf = (unsigned short*)walloc((size_t)BB * NN * HD * 2);
    (void)in_sizes; (void)n_in; (void)out_size; (void)ws_size;

    prep_kernel<<<512, 256, 0, stream>>>(x, Wq, Wk, Wv, Wo, xbf, WqT, WkT, WvT,
                                         WoT);

    const int nwaves = BB * HH * (NN / 16);   // 6144 waves, 8 per block
    qkv_kernel<<<nwaves / 8, 256, 0, stream>>>(xbf, WqT, WkT, WvT, bq, bk, bv,
                                               Qbf, Kbf, Vt);
    attn_kernel<<<nwaves / 8, 256, 0, stream>>>(Qbf, Kbf, Vt, Obf);

    const int pwaves = (BB * NN) / 16;        // 512 waves, 8 per block
    proj_kernel<<<pwaves / 8, 256, 0, stream>>>(Obf, WoT, bo, out);
}
